// DomainAdaptation_24000277250146
// MI455X (gfx1250) — matrix operand order repaired and checked by reading, not compiled, not run on hardware
//
#include <hip/hip_runtime.h>
#include <stdint.h>
#include <stddef.h>

// ---------------------------------------------------------------------------
// DomainAdaptation fused pipeline for MI455X (gfx1250, wave32, WMMA)
//
//   S      = Q^T K                    (bf16 WMMA, f32 accum)   [D,D]
//   P^T    = softmax(exp(S - S*I/32)) transposed, bf16         [D,D]
//   attn   = V @ P^T                  (bf16 WMMA) -> bf16      [N,D]
//   hid    = relu(attn@W1 + b1)       (bf16 WMMA) -> bf16      [N,H]
//   out    = hid@W2 + b2              (bf16 WMMA) -> f32       [N,D]   (x2: q,k)
//
// GEMM: block tile 128x64, BK=64, 8 waves (4Mx2N), 32x32 per wave ->
// 8 v_wmma_f32_16x16x32_bf16 per wave per LDS tile. A staged with
// global_load_async_to_lds_b128 (ASYNCcnt); B staged transposed via packed
// ds_store_b32.
// ---------------------------------------------------------------------------

typedef __bf16  v16bf __attribute__((ext_vector_type(16)));
typedef float   v8f   __attribute__((ext_vector_type(8)));

#define NSAMP 32768
#define DDIM  1024
#define HDIM  4096

__device__ __forceinline__ unsigned short f32_to_bf16_rne(float f) {
  union { float f; uint32_t u; } v; v.f = f;
  uint32_t u = v.u;
  u += 0x7FFFu + ((u >> 16) & 1u);          // round-to-nearest-even
  return (unsigned short)(u >> 16);
}

// ---------------------------------------------------------------- cast f32->bf16
__global__ __launch_bounds__(256) void cast_f32_bf16_kernel(
    const float* __restrict__ src, unsigned short* __restrict__ dst, int n4) {
  int i = blockIdx.x * 256 + threadIdx.x;
  if (i < n4) {
    float4 f = ((const float4*)src)[i];
    uint32_t lo = (uint32_t)f32_to_bf16_rne(f.x) | ((uint32_t)f32_to_bf16_rne(f.y) << 16);
    uint32_t hi = (uint32_t)f32_to_bf16_rne(f.z) | ((uint32_t)f32_to_bf16_rne(f.w) << 16);
    ((uint2*)dst)[i] = make_uint2(lo, hi);
  }
}

// ------------------------------------------------- softmax of exp-logits, transposed out
__global__ __launch_bounds__(256) void softmax_t_kernel(
    const float* __restrict__ S, unsigned short* __restrict__ PT) {
  const int i = blockIdx.x;          // row
  const int t = threadIdx.x;
  __shared__ float red[256];
  const float diag = 1.0f - 1.0f / 32.0f;   // 1 - 1/sqrt(1024)

  float tv[4];
  float lmax = -3.0e38f;
#pragma unroll
  for (int c = 0; c < 4; ++c) {
    int j = t + c * 256;
    float s = S[(size_t)i * DDIM + j];
    float x = expf(s * (j == i ? diag : 1.0f));   // logit = exp(...)
    tv[c] = x;
    lmax = fmaxf(lmax, x);
  }
  red[t] = lmax; __syncthreads();
  for (int s = 128; s > 0; s >>= 1) { if (t < s) red[t] = fmaxf(red[t], red[t + s]); __syncthreads(); }
  float rmax = red[0]; __syncthreads();

  float e[4]; float lsum = 0.f;
#pragma unroll
  for (int c = 0; c < 4; ++c) { e[c] = expf(tv[c] - rmax); lsum += e[c]; }
  red[t] = lsum; __syncthreads();
  for (int s = 128; s > 0; s >>= 1) { if (t < s) red[t] += red[t + s]; __syncthreads(); }
  float inv = 1.0f / red[0];

#pragma unroll
  for (int c = 0; c < 4; ++c) {
    int j = t + c * 256;
    PT[(size_t)j * DDIM + i] = f32_to_bf16_rne(e[c] * inv);   // transposed store
  }
}

// --------------------------------------------------------------- generic bf16 GEMM
enum { EPI_F32 = 0, EPI_BF16 = 1, EPI_BIAS_RELU_BF16 = 2, EPI_BIAS_F32 = 3 };

template <int EPI, bool TRANSA>
__global__ __launch_bounds__(256) void gemm_bf16_wmma(
    const unsigned short* __restrict__ A, const unsigned short* __restrict__ B,
    void* __restrict__ Cout, const float* __restrict__ bias,
    int M, int N, int K, int lda, int ldb, int ldc) {
  constexpr int BM = 128, BN = 64, BK = 64, PAD = 72;  // padded rows (144B) vs bank conflicts
  __shared__ __align__(16) unsigned short As[BM * PAD];     // [m][k]
  __shared__ __align__(16) unsigned short Bst[BN * PAD];    // [n][k]  (B staged transposed)

  const int tid  = threadIdx.x;
  const int lane = tid & 31;
  const int wave = tid >> 5;
  const int l16  = lane & 15;
  const int kh   = lane >> 4;            // lane half selects K sub-range
  const int wm   = (wave & 3) * 32;      // wave M offset inside block (4 waves over M)
  const int wn   = (wave >> 2) * 32;     // wave N offset inside block (2 waves over N)
  const int m0   = blockIdx.y * BM;
  const int n0   = blockIdx.x * BN;

  // LDS byte address of As for async-to-LDS (low 32 bits of flat addr == LDS offset)
  const uint32_t as_base = (uint32_t)(uintptr_t)(&As[0]);

  v8f acc00 = {0.f,0.f,0.f,0.f,0.f,0.f,0.f,0.f};
  v8f acc01 = {0.f,0.f,0.f,0.f,0.f,0.f,0.f,0.f};
  v8f acc10 = {0.f,0.f,0.f,0.f,0.f,0.f,0.f,0.f};
  v8f acc11 = {0.f,0.f,0.f,0.f,0.f,0.f,0.f,0.f};

  for (int k0 = 0; k0 < K; k0 += BK) {
    // ---- stage A tile into LDS As[m][k] ----
    if (!TRANSA) {
      // 128 rows x 64 ushorts; each thread DMAs 4x16B of one half-row, no VGPR data
      const int row  = tid >> 1;
      const int half = tid & 1;
      const unsigned short* gp = A + (size_t)(m0 + row) * lda + k0 + half * 32;
      const uint32_t lp = as_base + (uint32_t)(row * PAD + half * 32) * 2u;
#pragma unroll
      for (int i = 0; i < 4; ++i) {
        asm volatile("global_load_async_to_lds_b128 %0, %1, off"
                     :: "v"(lp + i * 16u), "v"(gp + i * 8) : "memory");
      }
    } else {
      // A mem is [K,M]; transpose 64k x 128m. Each thread: 4 k-rows x 8 m,
      // repacked as ushort4 (4 consecutive K) -> ds_store_b64.
      const int mseg = tid & 15;         // m chunk of 8
      const int kq   = tid >> 4;         // 0..15 -> k rows kq*4..+3
      const unsigned short* ap = A + (size_t)(k0 + kq * 4) * lda + m0 + mseg * 8;
      uint4 r0 = *(const uint4*)(ap);
      uint4 r1 = *(const uint4*)(ap + lda);
      uint4 r2 = *(const uint4*)(ap + 2 * lda);
      uint4 r3 = *(const uint4*)(ap + 3 * lda);
      const unsigned short* s0 = (const unsigned short*)&r0;
      const unsigned short* s1 = (const unsigned short*)&r1;
      const unsigned short* s2 = (const unsigned short*)&r2;
      const unsigned short* s3 = (const unsigned short*)&r3;
#pragma unroll
      for (int i = 0; i < 8; ++i) {
        uint2 v;
        v.x = (uint32_t)s0[i] | ((uint32_t)s1[i] << 16);
        v.y = (uint32_t)s2[i] | ((uint32_t)s3[i] << 16);
        *(uint2*)(&As[(mseg * 8 + i) * PAD + kq * 4]) = v;
      }
    }
    // ---- stage B tile transposed into LDS Bst[n][k] ----
    {
      // 64k x 64n. Each thread: 2 k-rows x 8 n, packed pairs -> ds_store_b32.
      const int nseg = tid & 7;          // n chunk of 8
      const int kg   = tid >> 3;         // 0..31 -> k rows kg*2, kg*2+1
      const unsigned short* bp = B + (size_t)(k0 + kg * 2) * ldb + n0 + nseg * 8;
      uint4 r0 = *(const uint4*)(bp);
      uint4 r1 = *(const uint4*)(bp + ldb);
      const unsigned short* s0 = (const unsigned short*)&r0;
      const unsigned short* s1 = (const unsigned short*)&r1;
#pragma unroll
      for (int i = 0; i < 8; ++i) {
        uint32_t v = (uint32_t)s0[i] | ((uint32_t)s1[i] << 16);
        *(uint32_t*)(&Bst[(nseg * 8 + i) * PAD + kg * 2]) = v;
      }
    }
    // prefetch next B tile (global_prefetch_b8)
    if (k0 + BK < K) {
      const int nseg = tid & 7, kg = tid >> 3;
      __builtin_prefetch(B + (size_t)(k0 + BK + kg * 2) * ldb + n0 + nseg * 8, 0, 3);
    }
    if (!TRANSA) {
      asm volatile("s_wait_asynccnt 0" ::: "memory");  // A DMA done before barrier
    }
    __syncthreads();

    // ---- 2 K-steps of 32; fragments per ISA 7.12.2 layouts ----
#pragma unroll
    for (int ks = 0; ks < 2; ++ks) {
      const int ko = ks * 32;
      union { v16bf v; uint4 q[2]; } a0, a1, b0, b1;
      // A 16x32: lane half 0 -> K 0..7 & 16..23 ; half 1 -> K 8..15 & 24..31
      const int ar0 = (wm + l16) * PAD + ko + kh * 8;
      a0.q[0] = *(const uint4*)(&As[ar0]);
      a0.q[1] = *(const uint4*)(&As[ar0 + 16]);
      const int ar1 = (wm + 16 + l16) * PAD + ko + kh * 8;
      a1.q[0] = *(const uint4*)(&As[ar1]);
      a1.q[1] = *(const uint4*)(&As[ar1 + 16]);
      // B 32x16: lane half 0 -> K 0..7 & 16..23 ; half 1 -> K 8..15 & 24..31
      const int br0 = (wn + l16) * PAD + ko + kh * 8;
      b0.q[0] = *(const uint4*)(&Bst[br0]);
      b0.q[1] = *(const uint4*)(&Bst[br0 + 16]);
      const int br1 = (wn + 16 + l16) * PAD + ko + kh * 8;
      b1.q[0] = *(const uint4*)(&Bst[br1]);
      b1.q[1] = *(const uint4*)(&Bst[br1 + 16]);

      acc00 = __builtin_amdgcn_wmma_f32_16x16x32_bf16(false, a0.v, false, b0.v, (short)0, acc00, false, false);
      acc01 = __builtin_amdgcn_wmma_f32_16x16x32_bf16(false, a0.v, false, b1.v, (short)0, acc01, false, false);
      acc10 = __builtin_amdgcn_wmma_f32_16x16x32_bf16(false, a1.v, false, b0.v, (short)0, acc10, false, false);
      acc11 = __builtin_amdgcn_wmma_f32_16x16x32_bf16(false, a1.v, false, b1.v, (short)0, acc11, false, false);
    }
    __syncthreads();
  }

  // ---- epilogue: C/D layout: col = lane&15, row = vgpr_j + 8*(lane>=16) ----
#pragma unroll
  for (int tm = 0; tm < 2; ++tm) {
    const int mbase = m0 + wm + tm * 16 + kh * 8;
#pragma unroll
    for (int j = 0; j < 8; ++j) {
      const int m = mbase + j;
#pragma unroll
      for (int tn = 0; tn < 2; ++tn) {
        const int n = n0 + wn + tn * 16 + l16;
        float v;
        if (tm == 0) v = (tn == 0) ? acc00[j] : acc01[j];
        else         v = (tn == 0) ? acc10[j] : acc11[j];
        if constexpr (EPI == EPI_F32) {
          ((float*)Cout)[(size_t)m * ldc + n] = v;
        } else if constexpr (EPI == EPI_BF16) {
          ((unsigned short*)Cout)[(size_t)m * ldc + n] = f32_to_bf16_rne(v);
        } else if constexpr (EPI == EPI_BIAS_RELU_BF16) {
          v += bias[n];
          v = v > 0.f ? v : 0.f;
          ((unsigned short*)Cout)[(size_t)m * ldc + n] = f32_to_bf16_rne(v);
        } else {  // EPI_BIAS_F32
          v += bias[n];
          ((float*)Cout)[(size_t)m * ldc + n] = v;
        }
      }
    }
  }
}

// ---------------------------------------------------------------------------
extern "C" void kernel_launch(void* const* d_in, const int* in_sizes, int n_in,
                              void* d_out, int out_size, void* d_ws, size_t ws_size,
                              hipStream_t stream) {
  (void)in_sizes; (void)n_in; (void)out_size; (void)ws_size;
  const float* q   = (const float*)d_in[0];
  const float* k   = (const float*)d_in[1];
  const float* v   = (const float*)d_in[2];
  const float* wq1 = (const float*)d_in[3];
  const float* bq1 = (const float*)d_in[4];
  const float* wq2 = (const float*)d_in[5];
  const float* bq2 = (const float*)d_in[6];
  const float* wk1 = (const float*)d_in[7];
  const float* bk1 = (const float*)d_in[8];
  const float* wk2 = (const float*)d_in[9];
  const float* bk2 = (const float*)d_in[10];

  char* ws = (char*)d_ws;
  size_t off = 0;
  auto carve = [&](size_t bytes) { char* p = ws + off; off = (off + bytes + 255) & ~(size_t)255; return p; };
  unsigned short* Qb   = (unsigned short*)carve((size_t)NSAMP * DDIM * 2);
  unsigned short* Kb   = (unsigned short*)carve((size_t)NSAMP * DDIM * 2);
  unsigned short* Vb   = (unsigned short*)carve((size_t)NSAMP * DDIM * 2);
  unsigned short* Wq1b = (unsigned short*)carve((size_t)DDIM * HDIM * 2);
  unsigned short* Wq2b = (unsigned short*)carve((size_t)HDIM * DDIM * 2);
  unsigned short* Wk1b = (unsigned short*)carve((size_t)DDIM * HDIM * 2);
  unsigned short* Wk2b = (unsigned short*)carve((size_t)HDIM * DDIM * 2);
  float*          Sf   = (float*)carve((size_t)DDIM * DDIM * 4);
  unsigned short* Pt   = (unsigned short*)carve((size_t)DDIM * DDIM * 2);
  unsigned short* Attn = (unsigned short*)carve((size_t)NSAMP * DDIM * 2);
  unsigned short* Hid  = (unsigned short*)carve((size_t)NSAMP * HDIM * 2);

  float* outq = (float*)d_out;
  float* outk = outq + (size_t)NSAMP * DDIM;

  auto cast = [&](const float* s, unsigned short* d, size_t n) {
    int n4 = (int)(n / 4);
    cast_f32_bf16_kernel<<<(n4 + 255) / 256, 256, 0, stream>>>(s, d, n4);
  };
  cast(q, Qb, (size_t)NSAMP * DDIM);
  cast(k, Kb, (size_t)NSAMP * DDIM);
  cast(v, Vb, (size_t)NSAMP * DDIM);
  cast(wq1, Wq1b, (size_t)DDIM * HDIM);
  cast(wq2, Wq2b, (size_t)HDIM * DDIM);
  cast(wk1, Wk1b, (size_t)DDIM * HDIM);
  cast(wk2, Wk2b, (size_t)HDIM * DDIM);

  dim3 blk(256);
  // S = Q^T K : M=N=D, K=NSAMP (A read transposed from Q[NSAMP,D])
  gemm_bf16_wmma<EPI_F32, true><<<dim3(DDIM / 64, DDIM / 128), blk, 0, stream>>>(
      Qb, Kb, Sf, nullptr, DDIM, DDIM, NSAMP, DDIM, DDIM, DDIM);
  softmax_t_kernel<<<DDIM, 256, 0, stream>>>(Sf, Pt);
  // attn = V @ P^T -> bf16
  gemm_bf16_wmma<EPI_BF16, false><<<dim3(DDIM / 64, NSAMP / 128), blk, 0, stream>>>(
      Vb, Pt, Attn, nullptr, NSAMP, DDIM, DDIM, DDIM, DDIM, DDIM);
  // dom_query MLP
  gemm_bf16_wmma<EPI_BIAS_RELU_BF16, false><<<dim3(HDIM / 64, NSAMP / 128), blk, 0, stream>>>(
      Attn, Wq1b, Hid, bq1, NSAMP, HDIM, DDIM, DDIM, HDIM, HDIM);
  gemm_bf16_wmma<EPI_BIAS_F32, false><<<dim3(DDIM / 64, NSAMP / 128), blk, 0, stream>>>(
      Hid, Wq2b, outq, bq2, NSAMP, DDIM, HDIM, HDIM, DDIM, DDIM);
  // dom_key MLP (reuse Hid)
  gemm_bf16_wmma<EPI_BIAS_RELU_BF16, false><<<dim3(HDIM / 64, NSAMP / 128), blk, 0, stream>>>(
      Attn, Wk1b, Hid, bk1, NSAMP, HDIM, DDIM, DDIM, HDIM, HDIM);
  gemm_bf16_wmma<EPI_BIAS_F32, false><<<dim3(DDIM / 64, NSAMP / 128), blk, 0, stream>>>(
      Hid, Wk2b, outk, bk2, NSAMP, DDIM, HDIM, HDIM, DDIM, DDIM);
}
